// GAT_22617297781051
// MI455X (gfx1250) — matrix operand order repaired and checked by reading, not compiled, not run on hardware
//
#include <hip/hip_runtime.h>
#include <hip/hip_bf16.h>
#include <math.h>

#define NODES 20000
#define NEG_SLOPE 0.2f

typedef __bf16 bf16_t;
typedef __attribute__((ext_vector_type(8)))  __bf16 v8bf;
typedef __attribute__((ext_vector_type(16))) __bf16 v16bf;
typedef __attribute__((ext_vector_type(8)))  float  v8f;

// ---------- helpers ----------
__device__ __forceinline__ bf16_t f32_to_bf16(float f) {
    unsigned u = __float_as_uint(f);
    unsigned r = u + 0x7FFFu + ((u >> 16) & 1u);        // round-to-nearest-even
    unsigned short s = (unsigned short)(r >> 16);
    bf16_t out;
    __builtin_memcpy(&out, &s, 2);
    return out;
}

__device__ __forceinline__ void atomicMaxF(float* addr, float val) {
    int* ia = (int*)addr;
    int old = __float_as_int(*addr);
    while (val > __int_as_float(old)) {
        int assumed = old;
        old = atomicCAS(ia, assumed, __float_as_int(val));
        if (old == assumed) break;
    }
}

// ---------- f32 -> bf16 convert ----------
__global__ void k_f2bf(const float* __restrict__ in, bf16_t* __restrict__ out, int n) {
    int i = blockIdx.x * blockDim.x + threadIdx.x;
    if (i < n) out[i] = f32_to_bf16(in[i]);
}

// ---------- pack W[K][M] (row-major f32) into WMMA B-fragment-major bf16 ----------
// Contiguous k order, same as A in k_gemm (32x16 bf16 B): lane l -> col = nt*16 + (l&15),
// element e (0..15) -> K = kt*32 + (l>>4)*16 + e.
// Stored so GEMM loads one contiguous 32B vector per lane:
//   Wp[ ((kt*nTiles+nt)*32 + lane)*16 + e ]
__global__ void k_pack_w(const float* __restrict__ W, bf16_t* __restrict__ Wp, int K, int M) {
    int idx = blockIdx.x * blockDim.x + threadIdx.x;
    int nTiles = M >> 4;
    int total = (K >> 5) * nTiles * 512;
    if (idx >= total) return;
    int e    = idx & 15;
    int lane = (idx >> 4) & 31;
    int tile = idx >> 9;               // kt*nTiles + nt
    int kt = tile / nTiles, nt = tile - kt * nTiles;
    int k = kt * 32 + (lane >> 4) * 16 + e;
    int n = nt * 16 + (lane & 15);
    Wp[idx] = f32_to_bf16(W[(size_t)k * M + n]);
}

// ---------- WMMA GEMM:  C[N][M] = A[N][K](bf16,row-major) x Wp(packed bf16) ----------
// Fully specialized: KSTEPS = K/32, NT column tiles/wave, MT row tiles/wave.
// block = 128 threads = 4 waves; each wave: (16*MT) rows x (16*NT) cols.
// No divergent/predicated code inside the K loop -> accumulators stay pinned.
template<int KSTEPS, int NT, int MT>
__global__ __launch_bounds__(128)
void k_gemm(const bf16_t* __restrict__ A, const bf16_t* __restrict__ Wp,
            float* __restrict__ C, int Nrows, int M) {
    constexpr int K = KSTEPS * 32;
    const int lane    = threadIdx.x & 31;
    const int wave    = threadIdx.x >> 5;
    const int nTiles  = M >> 4;
    const int rowBase = (blockIdx.y * 4 + wave) * (16 * MT);
    if (rowBase >= Nrows) return;                  // wave-uniform
    const int ntBase  = blockIdx.x * NT;
    const int half    = lane >> 4;
    const int laneRow = lane & 15;

    v8f acc[MT][NT] = {};

#pragma unroll
    for (int kt = 0; kt < KSTEPS; ++kt) {
        // A fragments: two contiguous 16B loads per lane (k = half*16 + e, matches packed B)
        v16bf a[MT];
#pragma unroll
        for (int mt = 0; mt < MT; ++mt) {
            const bf16_t* ap = A + (size_t)(rowBase + mt * 16 + laneRow) * K + kt * 32 + half * 16;
            v8bf lo = *(const v8bf*)(ap);
            v8bf hi = *(const v8bf*)(ap + 8);
            a[mt] = __builtin_shufflevector(lo, hi,
                    0, 1, 2, 3, 4, 5, 6, 7, 8, 9, 10, 11, 12, 13, 14, 15);
        }
#pragma unroll
        for (int t = 0; t < NT; ++t) {
            const bf16_t* bp = Wp + (((size_t)kt * nTiles + (ntBase + t)) * 32 + lane) * 16;
            v16bf b = *(const v16bf*)(bp);
#pragma unroll
            for (int mt = 0; mt < MT; ++mt) {
                acc[mt][t] = __builtin_amdgcn_wmma_f32_16x16x32_bf16(
                    false, a[mt], false, b, (short)0, acc[mt][t], false, false);
            }
        }
    }

    // C/D layout: lane l -> col = n0+(l&15); VGPR r -> row base + (l>>4)*8 + r
#pragma unroll
    for (int mt = 0; mt < MT; ++mt) {
#pragma unroll
        for (int t = 0; t < NT; ++t) {
            int col = (ntBase + t) * 16 + laneRow;
#pragma unroll
            for (int r = 0; r < 8; ++r) {
                int row = rowBase + mt * 16 + half * 8 + r;
                C[(size_t)row * M + col] = acc[mt][t][r];
            }
        }
    }
}

// ---------- per-node attention logits ----------
__global__ void k_logits(const float* __restrict__ h, const float* __restrict__ asr,
                         const float* __restrict__ adr, float* __restrict__ als,
                         float* __restrict__ ald, int Nn, int H, int C) {
    int idx = blockIdx.x * blockDim.x + threadIdx.x;
    if (idx >= Nn * H) return;
    int n = idx / H, hh = idx - n * H;
    const float* hp = h + ((size_t)n * H + hh) * C;
    float s = 0.f, d = 0.f;
    for (int c = 0; c < C; ++c) {
        float v = hp[c];
        s += v * asr[hh * C + c];
        d += v * adr[hh * C + c];
    }
    als[idx] = s;
    ald[idx] = d;
}

__global__ void k_fill(float* __restrict__ p, float v, int n) {
    int i = blockIdx.x * blockDim.x + threadIdx.x;
    if (i < n) p[i] = v;
}

// ---------- edge pass 1: leaky-relu logit + segment max ----------
__global__ void k_edge_max(const int* __restrict__ src, const int* __restrict__ dst,
                           int E, int Etot, const float* __restrict__ als,
                           const float* __restrict__ ald, float* __restrict__ lbuf,
                           float* __restrict__ emax, int H) {
    int idx = blockIdx.x * blockDim.x + threadIdx.x;
    if (idx >= Etot * H) return;
    int e = idx / H, hh = idx - e * H;
    int s = (e < E) ? src[e] : (e - E);     // self-loops appended
    int d = (e < E) ? dst[e] : (e - E);
    float l = als[s * H + hh] + ald[d * H + hh];
    l = (l > 0.f) ? l : NEG_SLOPE * l;
    lbuf[idx] = l;
    atomicMaxF(&emax[d * H + hh], l);
}

// ---------- edge pass 2: exp + segment sum ----------
__global__ void k_edge_exp(const int* __restrict__ dst, int E, int Etot,
                           float* __restrict__ lbuf, const float* __restrict__ emax,
                           float* __restrict__ den, int H) {
    int idx = blockIdx.x * blockDim.x + threadIdx.x;
    if (idx >= Etot * H) return;
    int e = idx / H, hh = idx - e * H;
    int d = (e < E) ? dst[e] : (e - E);
    float ee = __expf(lbuf[idx] - emax[d * H + hh]);
    lbuf[idx] = ee;
    atomicAdd(&den[d * H + hh], ee);
}

// ---------- edge pass 3: weighted message scatter ----------
// One thread per (edge, head, channel): consecutive lanes hit consecutive
// channels -> coalesced loads of h[src] and coalesced atomic bursts into L2.
__global__ void k_edge_aggr(const int* __restrict__ src, const int* __restrict__ dst,
                            int E, int Etot, const float* __restrict__ lbuf,
                            const float* __restrict__ den, const float* __restrict__ h,
                            float* __restrict__ acc, int H, int C, int logC) {
    int idx = blockIdx.x * blockDim.x + threadIdx.x;
    if (idx >= Etot * H * C) return;                 // max 87.04M < 2^31
    int c  = idx & (C - 1);
    int eh = idx >> logC;                            // e*H + hh
    int e  = eh / H, hh = eh - e * H;
    int s = (e < E) ? src[e] : (e - E);
    int d = (e < E) ? dst[e] : (e - E);
    float coef = lbuf[eh] / den[d * H + hh];
    float val  = coef * h[((size_t)s * H + hh) * C + c];
    atomicAdd(&acc[((size_t)d * H + hh) * C + c], val);
}

// ---------- BatchNorm stats ----------
__global__ __launch_bounds__(256)
void k_bn_stats(const float* __restrict__ acc, const float* __restrict__ bias,
                const float* __restrict__ gamma, const float* __restrict__ beta,
                float* __restrict__ scale, float* __restrict__ shift, int Nn, int Ch) {
    int ch = blockIdx.x;
    __shared__ float ss[256], sq[256];
    float s = 0.f, q = 0.f;
    float b = bias[ch];
    for (int n = threadIdx.x; n < Nn; n += 256) {
        float v = acc[(size_t)n * Ch + ch] + b;
        s += v; q += v * v;
    }
    ss[threadIdx.x] = s; sq[threadIdx.x] = q;
    __syncthreads();
    for (int st = 128; st > 0; st >>= 1) {
        if (threadIdx.x < st) { ss[threadIdx.x] += ss[threadIdx.x + st]; sq[threadIdx.x] += sq[threadIdx.x + st]; }
        __syncthreads();
    }
    if (threadIdx.x == 0) {
        float mean = ss[0] / Nn;
        float var  = sq[0] / Nn - mean * mean;
        float sc   = gamma[ch] * rsqrtf(var + 1e-5f);
        scale[ch] = sc;
        shift[ch] = beta[ch] - mean * sc;
    }
}

// ---------- BN apply + ReLU, emit bf16 input for next layer's GEMM ----------
__global__ void k_bn_apply(const float* __restrict__ acc, const float* __restrict__ bias,
                           const float* __restrict__ scale, const float* __restrict__ shift,
                           bf16_t* __restrict__ out, int total, int Ch) {
    int idx = blockIdx.x * blockDim.x + threadIdx.x;
    if (idx >= total) return;
    int ch = idx % Ch;
    float v = (acc[idx] + bias[ch]) * scale[ch] + shift[ch];
    out[idx] = f32_to_bf16(fmaxf(v, 0.f));
}

// ---------- final: +bias, log_softmax over 16 channels ----------
__global__ void k_final(const float* __restrict__ acc, const float* __restrict__ b2,
                        float* __restrict__ out, int Nn) {
    int n = blockIdx.x * blockDim.x + threadIdx.x;
    if (n >= Nn) return;
    float v[16], m = -1e30f;
    for (int c = 0; c < 16; ++c) {
        v[c] = acc[(size_t)n * 16 + c] + b2[c];
        m = fmaxf(m, v[c]);
    }
    float ssum = 0.f;
    for (int c = 0; c < 16; ++c) ssum += __expf(v[c] - m);
    float ls = logf(ssum);
    for (int c = 0; c < 16; ++c) out[(size_t)n * 16 + c] = v[c] - m - ls;
}

// ============================ driver ============================
extern "C" void kernel_launch(void* const* d_in, const int* in_sizes, int n_in,
                              void* d_out, int out_size, void* d_ws, size_t ws_size,
                              hipStream_t stream) {
    const float* x   = (const float*)d_in[0];
    const int*   ei  = (const int*)  d_in[1];
    const float* W0  = (const float*)d_in[2];
    const float* as0 = (const float*)d_in[3];
    const float* ad0 = (const float*)d_in[4];
    const float* b0  = (const float*)d_in[5];
    const float* g0  = (const float*)d_in[6];
    const float* be0 = (const float*)d_in[7];
    const float* W1  = (const float*)d_in[8];
    const float* as1 = (const float*)d_in[9];
    const float* ad1 = (const float*)d_in[10];
    const float* b1  = (const float*)d_in[11];
    const float* g1  = (const float*)d_in[12];
    const float* be1 = (const float*)d_in[13];
    const float* W2  = (const float*)d_in[14];
    const float* as2 = (const float*)d_in[15];
    const float* ad2 = (const float*)d_in[16];
    const float* b2  = (const float*)d_in[17];

    const int E    = in_sizes[1] / 2;
    const int Etot = E + NODES;
    const int* srcv = ei;
    const int* dstv = ei + E;

    // workspace carve-out (256B aligned)
    char* base = (char*)d_ws; size_t cur = 0;
    auto take = [&](size_t bytes) -> char* {
        cur = (cur + 255) & ~(size_t)255;
        char* p = base + cur; cur += bytes; return p;
    };
    bf16_t* Abf  = (bf16_t*)take((size_t)NODES * 256 * 2);   // GEMM input (bf16)
    bf16_t* Wp   = (bf16_t*)take((size_t)8 * 16 * 512 * 2);  // packed W (max 128KB)
    float*  h    = (float*) take((size_t)NODES * 256 * 4);   // projected features
    float*  acc  = (float*) take((size_t)NODES * 256 * 4);   // aggregation accumulator
    float*  als  = (float*) take((size_t)NODES * 8 * 4);
    float*  ald  = (float*) take((size_t)NODES * 8 * 4);
    float*  emax = (float*) take((size_t)NODES * 8 * 4);
    float*  den  = (float*) take((size_t)NODES * 8 * 4);
    float*  lbuf = (float*) take((size_t)Etot  * 8 * 4);     // edge logits/alpha
    float*  scale= (float*) take(256 * 4);
    float*  shift= (float*) take(256 * 4);

    // 20000 rows = 625 wave-rows of 32; 4 waves/block -> 157 blocks in y
    const dim3 gemmGridBig(4, 157);   // M=256: 16 col tiles / 4 per wave
    const dim3 gemmGridSmall(1, 157); // M=16:  1 col tile

    auto attention = [&](const float* asr, const float* adr, int H, int C, int logC) {
        int nh = NODES * H;
        k_logits<<<(nh + 255) / 256, 256, 0, stream>>>(h, asr, adr, als, ald, NODES, H, C);
        k_fill<<<(nh + 255) / 256, 256, 0, stream>>>(emax, -1e30f, nh);
        k_fill<<<(nh + 255) / 256, 256, 0, stream>>>(den, 0.f, nh);
        int nm = NODES * H * C;
        k_fill<<<(nm + 255) / 256, 256, 0, stream>>>(acc, 0.f, nm);

        int eh = Etot * H;
        k_edge_max<<<(eh + 255) / 256, 256, 0, stream>>>(srcv, dstv, E, Etot, als, ald, lbuf, emax, H);
        k_edge_exp<<<(eh + 255) / 256, 256, 0, stream>>>(dstv, E, Etot, lbuf, emax, den, H);
        int ehc = Etot * H * C;
        k_edge_aggr<<<(ehc + 255) / 256, 256, 0, stream>>>(srcv, dstv, E, Etot, lbuf, den, h, acc, H, C, logC);
    };

    // ---- layer 0: K=128, M=256, H=8, C=32 ----
    k_f2bf<<<(NODES * 128 + 255) / 256, 256, 0, stream>>>(x, Abf, NODES * 128);
    k_pack_w<<<((128 >> 5) * 16 * 512 + 255) / 256, 256, 0, stream>>>(W0, Wp, 128, 256);
    k_gemm<4, 4, 2><<<gemmGridBig, 128, 0, stream>>>(Abf, Wp, h, NODES, 256);
    attention(as0, ad0, 8, 32, 5);
    k_bn_stats<<<256, 256, 0, stream>>>(acc, b0, g0, be0, scale, shift, NODES, 256);
    k_bn_apply<<<(NODES * 256 + 255) / 256, 256, 0, stream>>>(acc, b0, scale, shift, Abf, NODES * 256, 256);

    // ---- layer 1: K=256, M=256, H=8, C=32 ----
    k_pack_w<<<((256 >> 5) * 16 * 512 + 255) / 256, 256, 0, stream>>>(W1, Wp, 256, 256);
    k_gemm<8, 4, 2><<<gemmGridBig, 128, 0, stream>>>(Abf, Wp, h, NODES, 256);
    attention(as1, ad1, 8, 32, 5);
    k_bn_stats<<<256, 256, 0, stream>>>(acc, b1, g1, be1, scale, shift, NODES, 256);
    k_bn_apply<<<(NODES * 256 + 255) / 256, 256, 0, stream>>>(acc, b1, scale, shift, Abf, NODES * 256, 256);

    // ---- layer 2: K=256, M=16, H=1, C=16 (mean over 1 head == identity) ----
    k_pack_w<<<((256 >> 5) * 1 * 512 + 255) / 256, 256, 0, stream>>>(W2, Wp, 256, 16);
    k_gemm<8, 1, 2><<<gemmGridSmall, 128, 0, stream>>>(Abf, Wp, h, NODES, 16);
    attention(as2, ad2, 1, 16, 4);
    k_final<<<(NODES + 255) / 256, 256, 0, stream>>>(acc, b2, (float*)d_out, NODES);
}
